// RNN_54863912239217
// MI455X (gfx1250) — compile-verified, matrix operand order checked by reading, not run on hardware
//
#include <hip/hip_runtime.h>

// GRU (H=32, I=1, B=2048, T=1024) on gfx1250.
// One wave32 = one 16-batch-row tile; per timestep: 6x v_wmma_f32_16x16x32_f16
// (M=16, N=96 gates, K=32=H), LDS transpose round-trip for the h D->A fragment
// relayout, v_tanh_f32 activations, shfl reduction for the Linear(32,1) output.

typedef __attribute__((ext_vector_type(16))) _Float16 v16h;
typedef __attribute__((ext_vector_type(8)))  _Float16 v8h;
typedef __attribute__((ext_vector_type(8)))  float    v8f;

#define B_SZ 2048
#define T_SZ 1024
#define H_SZ 32
#define CH   32   // x staging chunk (timesteps)
#define XSTR 20   // padded LDS row stride (floats) for the x chunk

__device__ __forceinline__ float fast_tanh(float x) {
#if defined(__has_builtin) && __has_builtin(__builtin_amdgcn_tanhf)
  return __builtin_amdgcn_tanhf(x);
#else
  float r;
  asm volatile("v_tanh_f32 %0, %1" : "=v"(r) : "v"(x));
  return r;
#endif
}
__device__ __forceinline__ float fast_sigmoid(float x) {
  return 0.5f * fast_tanh(0.5f * x) + 0.5f;
}

__global__ __launch_bounds__(128, 1) void gru_wmma_gfx1250(
    const float* __restrict__ x,    // [B,T,1]
    const float* __restrict__ h0,   // [1,B,H]
    const float* __restrict__ Wih,  // [3H,1]
    const float* __restrict__ Whh,  // [3H,H]
    const float* __restrict__ bih,  // [3H]
    const float* __restrict__ bhh,  // [3H]
    const float* __restrict__ Wout, // [1,H]
    const float* __restrict__ bout, // [1]
    float* __restrict__ y,          // [B,T]
    float* __restrict__ hn)         // [B,H]
{
  __shared__ alignas(16) _Float16 hbuf[4][16 * H_SZ];   // per-wave transpose tile
  __shared__ alignas(16) float    xls[4][CH * XSTR];    // per-wave x chunk [t][row]

  const int lane = threadIdx.x & 31;
  const int wv   = threadIdx.x >> 5;
  const int lo   = lane & 15;   // column / N-lane within half-wave
  const int hf   = lane >> 4;   // half-wave select
  const int tile = blockIdx.x * 4 + wv;
  const int row0 = tile * 16;   // first batch row of this wave's tile

  // ---- W_hh -> 6 B-fragments (f16), gate columns n = tt*16 + lo ----
  // B-frag (32x16, 16-bit): lane n, VGPR v holds K = {2v (+8 if v>=4)} + 8*hf, pair.
  v16h bW[6];
#pragma unroll
  for (int tt = 0; tt < 6; ++tt) {
    const int g = tt * 16 + lo;
#pragma unroll
    for (int v = 0; v < 8; ++v) {
      const int kb = ((v < 4) ? (2 * v) : (2 * v + 8)) + 8 * hf;
      bW[tt][2 * v]     = (_Float16)Whh[g * H_SZ + kb];
      bW[tt][2 * v + 1] = (_Float16)Whh[g * H_SZ + kb + 1];
    }
  }

  // ---- per-lane gate constants ----
  float wih_rz[4], cini[4];           // r/z tiles: fold b_ih + b_hh into C init
#pragma unroll
  for (int tt = 0; tt < 4; ++tt) {
    const int g = tt * 16 + lo;
    wih_rz[tt] = Wih[g];
    cini[tt]   = bih[g] + bhh[g];
  }
  float wih_n[2], bih_n[2], bhh_n[2]; // n tiles: gh_n kept separate (r * gh_n)
#pragma unroll
  for (int jt = 0; jt < 2; ++jt) {
    const int g = (4 + jt) * 16 + lo;
    wih_n[jt] = Wih[g];
    bih_n[jt] = bih[g];
    bhh_n[jt] = bhh[g];
  }
  const float wo_a = Wout[lo], wo_b = Wout[16 + lo], bo = bout[0];

  // ---- hidden state f32, D-fragment layout: h[jt*8+r] = h[row0+r+8*hf][jt*16+lo]
  float h[16];
#pragma unroll
  for (int jt = 0; jt < 2; ++jt)
#pragma unroll
    for (int r = 0; r < 8; ++r)
      h[jt * 8 + r] = h0[(size_t)(row0 + r + 8 * hf) * H_SZ + jt * 16 + lo];

  for (int t = 0; t < T_SZ; ++t) {
    const int tc = t & (CH - 1);
    if (tc == 0) {
      // stage next 32 timesteps of x, coalesced along t; LDS layout [t][row]
#pragma unroll
      for (int i = 0; i < 16; ++i)
        xls[wv][lane * XSTR + i] = x[(size_t)(row0 + i) * T_SZ + t + lane];
      if (t + CH < T_SZ)
        __builtin_prefetch(&x[(size_t)(row0 + lo) * T_SZ + t + CH + hf * 16], 0, 1);
    }

    // ---- h (D-layout f32) -> f16 -> LDS -> A fragment (transpose) ----
#pragma unroll
    for (int jt = 0; jt < 2; ++jt)
#pragma unroll
      for (int r = 0; r < 8; ++r)
        hbuf[wv][(r + 8 * hf) * H_SZ + jt * 16 + lo] = (_Float16)h[jt * 8 + r];

    const int k0 = 8 * hf;  // A-frag: lane row = lo, K = k0..k0+7 and k0+16..k0+23
    v8h alo = *(const v8h*)&hbuf[wv][lo * H_SZ + k0];
    v8h ahi = *(const v8h*)&hbuf[wv][lo * H_SZ + k0 + 16];
    v16h aH = __builtin_shufflevector(alo, ahi,
        0, 1, 2, 3, 4, 5, 6, 7, 8, 9, 10, 11, 12, 13, 14, 15);

    // x values for this lane's 8 batch rows (r + 8*hf)
    const float4 xa = *(const float4*)&xls[wv][tc * XSTR + 8 * hf];
    const float4 xb = *(const float4*)&xls[wv][tc * XSTR + 8 * hf + 4];
    const float xv[8] = {xa.x, xa.y, xa.z, xa.w, xb.x, xb.y, xb.z, xb.w};

    // ---- accumulators: r/z seeded with xp+biases, n seeded with b_hh only ----
    v8f acc[6];
#pragma unroll
    for (int tt = 0; tt < 4; ++tt)
#pragma unroll
      for (int r = 0; r < 8; ++r)
        acc[tt][r] = xv[r] * wih_rz[tt] + cini[tt];
#pragma unroll
    for (int jt = 0; jt < 2; ++jt)
#pragma unroll
      for (int r = 0; r < 8; ++r)
        acc[4 + jt][r] = bhh_n[jt];

#pragma unroll
    for (int tt = 0; tt < 6; ++tt)
      acc[tt] = __builtin_amdgcn_wmma_f32_16x16x32_f16(
          false, aH, false, bW[tt], (short)0, acc[tt], false, false);

    // ---- gates + state update (all per-lane; v_tanh_f32) ----
#pragma unroll
    for (int jt = 0; jt < 2; ++jt)
#pragma unroll
      for (int r = 0; r < 8; ++r) {
        const float rg  = fast_sigmoid(acc[jt][r]);
        const float zg  = fast_sigmoid(acc[2 + jt][r]);
        const float xpn = xv[r] * wih_n[jt] + bih_n[jt];
        const float ng  = fast_tanh(xpn + rg * acc[4 + jt][r]);
        h[jt * 8 + r]   = (1.0f - zg) * ng + zg * h[jt * 8 + r];
      }

    // ---- y[b][t] = h_new . W_out + b_out : reduce over 16 lanes per half ----
#pragma unroll
    for (int r = 0; r < 8; ++r) {
      float p = h[r] * wo_a + h[8 + r] * wo_b;
      p += __shfl_xor(p, 1, 16);
      p += __shfl_xor(p, 2, 16);
      p += __shfl_xor(p, 4, 16);
      p += __shfl_xor(p, 8, 16);
      if (lo == 0)
        y[(size_t)(row0 + r + 8 * hf) * T_SZ + t] = p + bo;
    }
  }

  // ---- final hidden state ----
#pragma unroll
  for (int jt = 0; jt < 2; ++jt)
#pragma unroll
    for (int r = 0; r < 8; ++r)
      hn[(size_t)(row0 + r + 8 * hf) * H_SZ + jt * 16 + lo] = h[jt * 8 + r];
}

extern "C" void kernel_launch(void* const* d_in, const int* in_sizes, int n_in,
                              void* d_out, int out_size, void* d_ws, size_t ws_size,
                              hipStream_t stream) {
  (void)in_sizes; (void)n_in; (void)out_size; (void)d_ws; (void)ws_size;
  const float* x    = (const float*)d_in[0];
  const float* h0   = (const float*)d_in[1];
  const float* Wih  = (const float*)d_in[2];
  const float* Whh  = (const float*)d_in[3];
  const float* bih  = (const float*)d_in[4];
  const float* bhh  = (const float*)d_in[5];
  const float* Wout = (const float*)d_in[6];
  const float* bout = (const float*)d_in[7];
  float* y  = (float*)d_out;                    // [B,T] flat
  float* hn = y + (size_t)B_SZ * T_SZ;          // then [B,H] flat

  dim3 grid(B_SZ / (16 * 4));  // 32 blocks x 4 waves = 128 tiles of 16 rows
  dim3 block(128);
  hipLaunchKernelGGL(gru_wmma_gfx1250, grid, block, 0, stream,
                     x, h0, Wih, Whh, bih, bhh, Wout, bout, y, hn);
}
